// CrossAttnModule_4655744549558
// MI455X (gfx1250) — compile-verified, matrix operand order checked by reading, not run on hardware
//
#include <hip/hip_runtime.h>
#include <hip/hip_bf16.h>

// ---------------------------------------------------------------------------
// CDNA5 (gfx1250) fused cross-attention block.
//  - all matmuls on v_wmma_f32_16x16x32_bf16 (wave32)
//  - bf16 operands pre-converted; weights pre-transposed to [N][K] so every
//    GEMM tile stage is a contiguous 16B/lane copy
//  - global->LDS staging uses global_load_async_to_lds_b128 (double buffered);
//    attention kp/vp tiles pulled in by the Tensor Data Mover
// ---------------------------------------------------------------------------

typedef __attribute__((ext_vector_type(16))) __bf16 v16bf;
typedef __attribute__((ext_vector_type(8)))  __bf16 v8bf;
typedef __attribute__((ext_vector_type(8)))  float  v8f;
// GCC-style vectors to match the builtin declarations exactly
typedef int          v4i_g  __attribute__((vector_size(16)));
typedef unsigned int u32x4g __attribute__((vector_size(16)));
typedef int          i32x8g __attribute__((vector_size(32)));
typedef int          i32x4g __attribute__((vector_size(16)));

#if defined(__AMDGCN__) && __has_builtin(__builtin_amdgcn_tensor_load_to_lds)
#define HAS_TDM 1
#else
#define HAS_TDM 0
#endif
#if defined(__AMDGCN__) && __has_builtin(__builtin_amdgcn_global_load_async_to_lds_b128)
#define HAS_ASYNC 1
#else
#define HAS_ASYNC 0
#endif

static __device__ __forceinline__ v8f wmma_bf16(v16bf a, v16bf b, v8f c) {
  return __builtin_amdgcn_wmma_f32_16x16x32_bf16(false, a, false, b, (short)0, c,
                                                 false, false);
}
static __device__ __forceinline__ v16bf cat8(v8bf lo, v8bf hi) {
  return __builtin_shufflevector(lo, hi, 0, 1, 2, 3, 4, 5, 6, 7,
                                 8, 9, 10, 11, 12, 13, 14, 15);
}
static __device__ __forceinline__ v8bf ld8(const float* p) {
  const float4 a = ((const float4*)p)[0];
  const float4 b = ((const float4*)p)[1];
  v8bf r;
  r[0] = (__bf16)a.x; r[1] = (__bf16)a.y; r[2] = (__bf16)a.z; r[3] = (__bf16)a.w;
  r[4] = (__bf16)b.x; r[5] = (__bf16)b.y; r[6] = (__bf16)b.z; r[7] = (__bf16)b.w;
  return r;
}

// 16-byte global -> LDS copy: async DMA when available, else through VGPRs.
static __device__ __forceinline__ void cp16(v8bf* dst, const __bf16* src) {
#if HAS_ASYNC
  __builtin_amdgcn_global_load_async_to_lds_b128(
      (__attribute__((address_space(1))) v4i_g*)src,
      (__attribute__((address_space(3))) v4i_g*)dst, 0, 0);
#else
  *dst = *(const v8bf*)src;
#endif
}
static __device__ __forceinline__ void wait_async() {
#if HAS_ASYNC
  asm volatile("s_wait_asynccnt 0" ::: "memory");
#endif
}

#if HAS_TDM
// Linear block copy via the Tensor Data Mover (one wave issues, TENSORcnt).
static __device__ __forceinline__ void tdm_copy_lds(void* lds, const void* gsrc,
                                                    unsigned bytes) {
  const unsigned lds_a =
      (unsigned)(size_t)(__attribute__((address_space(3))) void*)lds;
  const unsigned long long ga = (unsigned long long)(size_t)gsrc;
  const unsigned words = bytes >> 2;  // 4-byte elements
  u32x4g g0;
  g0[0] = 1u;                                    // count=1, user descriptor
  g0[1] = lds_a;                                 // lds_addr
  g0[2] = (unsigned)(ga & 0xffffffffu);          // global_addr[31:0]
  g0[3] = (unsigned)((ga >> 32) & 0x01ffffffu)   // global_addr[56:32]
          | (2u << 30);                          // type = 2 ("image")
  i32x8g g1;
  g1[0] = (int)(2u << 16);                       // data_size = 4B
  g1[1] = (int)((words & 0xffffu) << 16);        // tensor_dim0[15:0]
  g1[2] = (int)((words >> 16) & 0xffffu) | (1 << 16);  // dim0 hi | tensor_dim1=1
  g1[3] = (int)((words & 0xffffu) << 16);        // tile_dim0
  g1[4] = 1;                                     // tile_dim1 = 1
  g1[5] = (int)words;                            // tensor_dim0_stride
  g1[6] = 0;
  g1[7] = 0;
  const i32x4g gz4 = {0, 0, 0, 0};
  const i32x8g gz8 = {0, 0, 0, 0, 0, 0, 0, 0};
  __builtin_amdgcn_tensor_load_to_lds(g0, g1, gz4, gz4, gz8, 0);
}
#endif

#define BB  8
#define NN1 4096
#define NN2 4096
#define CC  512
#define PP  256
#define HH  8
#define HD  64
#define MLPD 2048

// ---------------------------------------------------------------------------
// elementwise fp32 -> bf16 (rows kept)
// ---------------------------------------------------------------------------
__launch_bounds__(256)
__global__ void cvt_bf16(const float* __restrict__ src, __bf16* __restrict__ dst,
                         int n) {
  const int i = (blockIdx.x * 256 + threadIdx.x) * 8;
  if (i < n) *(v8bf*)(dst + i) = ld8(src + i);
}

// transpose + convert: dst[n*K + k] = (bf16) src[k*N + n]
__launch_bounds__(256)
__global__ void transpose_cvt(const float* __restrict__ src, __bf16* __restrict__ dst,
                              int K, int N) {
  const int idx = blockIdx.x * 256 + threadIdx.x;
  if (idx < K * N) {
    const int k = idx / N, n = idx - k * N;
    dst[(size_t)n * K + k] = (__bf16)src[idx];
  }
}

// ---------------------------------------------------------------------------
// GEMM: C[M,N] = A[M,K] @ B[K,N] + epilogue.  A bf16 [M][K], BT bf16 [N][K].
// BM=128,BN=64,BK=32; 256 thr = 8 waves; double-buffered async LDS staging.
// ---------------------------------------------------------------------------
enum { MODE_Q = 0, MODE_KV = 1, MODE_MLP1 = 2, MODE_MLP2 = 3 };

template <int MODE>
__launch_bounds__(256)
__global__ void gemm_wmma(const __bf16* __restrict__ A, const __bf16* __restrict__ BT,
                          const float* __restrict__ bias,
                          const float* __restrict__ aux,  // pos (Q/KV) / residual (MLP2)
                          float*  __restrict__ Cf,
                          __bf16* __restrict__ Cb0, __bf16* __restrict__ Cb1,
                          int M, int N, int K, int seqmod) {
  __shared__ v8bf As[2][128][4];   // [buf][m][kchunk]
  __shared__ v8bf Bst[2][64][4];   // [buf][n][kchunk] (B transposed)

  const int t = threadIdx.x;
  const int wave = t >> 5, lane = t & 31;
  const int nl = lane & 15, hsel = lane >> 4;
  const int m0 = blockIdx.y * 128, n0 = blockIdx.x * 64;

  const int kcA = t & 3, mA = t >> 2;   // A: chunks (mA, mA+64)
  const int nB = t & 63, kcB = t >> 6;  // B: one chunk

  v8f acc[4] = {};

  auto stage = [&](int buf, int k0) {
    cp16(&As[buf][mA][kcA],      A + (size_t)(m0 + mA) * K + k0 + kcA * 8);
    cp16(&As[buf][mA + 64][kcA], A + (size_t)(m0 + mA + 64) * K + k0 + kcA * 8);
    cp16(&Bst[buf][nB][kcB],     BT + (size_t)(n0 + nB) * K + k0 + kcB * 8);
  };

  stage(0, 0);
  wait_async();
  __syncthreads();

  const int nk = K >> 5;
  for (int ik = 0; ik < nk; ++ik) {
    const int cur = ik & 1;
    if (ik + 1 < nk) stage(cur ^ 1, (ik + 1) << 5);

    const int row = wave * 16 + nl;
    const v16bf a = cat8(As[cur][row][hsel], As[cur][row][2 + hsel]);
#pragma unroll
    for (int nt = 0; nt < 4; ++nt) {
      const int col = nt * 16 + nl;
      const v16bf b = cat8(Bst[cur][col][hsel], Bst[cur][col][2 + hsel]);
      acc[nt] = wmma_bf16(a, b, acc[nt]);
    }
    wait_async();
    __syncthreads();
  }

  const int rbase = wave * 16 + hsel * 8;

  if (MODE == MODE_KV) {
    // k/v split is uniform per block (n0 multiple of 64, boundary at 512).
    const bool isK = (n0 < (N >> 1));
    __bf16* __restrict__ dst = isK ? Cb0 : Cb1;   // layout [b][h][d][NN2]
    const int bidx = m0 >> 12;                    // batch (N2 = 4096)
    const int nlo = (m0 & (NN2 - 1)) + rbase;     // sequence pos of r=0
#pragma unroll
    for (int nt = 0; nt < 4; ++nt) {
      const int gn = n0 + nt * 16 + nl;
      const int c = gn & (CC - 1);
      const float bv = bias[gn];
      v8bf pk;
#pragma unroll
      for (int r = 0; r < 8; ++r) {
        float v = acc[nt][r] + bv;
        if (isK) v += aux[(size_t)(nlo + r) * CC + c];  // pos_k
        pk[r] = (__bf16)v;
      }
      *(v8bf*)&dst[((size_t)(bidx * HH + (c >> 6)) * HD + (c & 63)) * NN2 + nlo] = pk;
    }
  } else {
#pragma unroll
    for (int nt = 0; nt < 4; ++nt) {
      const int gn = n0 + nt * 16 + nl;
      const float bv = bias[gn];
#pragma unroll
      for (int r = 0; r < 8; ++r) {
        const int gm = m0 + rbase + r;
        float v = acc[nt][r] + bv;
        if (MODE == MODE_Q) {
          v += aux[(size_t)(gm % seqmod) * N + gn];
          Cf[(size_t)gm * N + gn] = v;
        } else if (MODE == MODE_MLP1) {
          const float g = 0.5f * v * (1.0f + erff(v * 0.70710678118654752f));
          Cb0[(size_t)gm * N + gn] = (__bf16)g;
        } else {  // MODE_MLP2
          v += aux[(size_t)gm * N + gn];
          Cf[(size_t)gm * N + gn] = v;
        }
      }
    }
  }
}

// ---------------------------------------------------------------------------
// kp/vp per (b,h): proj[d,p] = sum_n XT[bh][d][n]*Wp[n,p] + bp[p]  (64x256)
// A = XT tile (contiguous rows), B = WpT tile (contiguous rows).
// Outputs: kpT [bh][p][d] and vp [bh][d][p] (A-operand layouts for attention).
// ---------------------------------------------------------------------------
__launch_bounds__(256)
__global__ void kpvp_wmma(const __bf16* __restrict__ kT, const __bf16* __restrict__ vT,
                          const __bf16* __restrict__ WpT, const float* __restrict__ bp,
                          __bf16* __restrict__ kpT, __bf16* __restrict__ vp) {
  __shared__ v8bf As[64][4];
  __shared__ v8bf Bst[64][4];

  const int t = threadIdx.x, wave = t >> 5, lane = t & 31;
  const int nl = lane & 15, hsel = lane >> 4;
  const int bh = blockIdx.x;
  const int p0 = blockIdx.y * 64;
  const bool isV = (blockIdx.z != 0);
  const __bf16* __restrict__ srcT = isV ? vT : kT;

  const int mt = wave & 3;
  const int ntA = (wave >> 2) * 2;
  const int dA = t & 63, kcA = t >> 6;
  v8f acc[2] = {};

  for (int k0 = 0; k0 < NN2; k0 += 32) {
    cp16(&As[dA][kcA],  srcT + ((size_t)bh * HD + dA) * NN2 + k0 + kcA * 8);
    cp16(&Bst[dA][kcA], WpT + (size_t)(p0 + dA) * NN2 + k0 + kcA * 8);
    wait_async();
    __syncthreads();

    const int row = mt * 16 + nl;
    const v16bf a = cat8(As[row][hsel], As[row][2 + hsel]);
#pragma unroll
    for (int u = 0; u < 2; ++u) {
      const int col = (ntA + u) * 16 + nl;
      const v16bf bb = cat8(Bst[col][hsel], Bst[col][2 + hsel]);
      acc[u] = wmma_bf16(a, bb, acc[u]);
    }
    __syncthreads();
  }

  const int rbase = mt * 16 + hsel * 8;
#pragma unroll
  for (int u = 0; u < 2; ++u) {
    const int p = p0 + (ntA + u) * 16 + nl;
    const float bv = bp[p];
#pragma unroll
    for (int r = 0; r < 8; ++r) {
      const int d = rbase + r;
      const float v = acc[u][r] + bv;
      if (isV) vp [((size_t)bh * HD + d) * PP + p] = (__bf16)v;
      else     kpT[((size_t)bh * PP + p) * HD + d] = (__bf16)v;
    }
  }
}

// ---------------------------------------------------------------------------
// Fused transposed attention:
//   S^T = kpT @ q^T ; lane-local softmax (+1 shuffle) ; X^T = vp @ P^T where
//   P^T B-fragments are the S^T accumulators (register identity).
// kp/vp tiles pulled into LDS by the Tensor Data Mover.
// ---------------------------------------------------------------------------
__launch_bounds__(256)
__global__ void attn_fused(const float* __restrict__ q, const __bf16* __restrict__ kpT,
                           const __bf16* __restrict__ vp, const float* __restrict__ temp,
                           float* __restrict__ xo) {
  __shared__ v8bf kps[256][8];   // [p][dchunk]
  __shared__ v8bf vps[64][32];   // [d][pchunk]

  const int t = threadIdx.x, wave = t >> 5, lane = t & 31;
  const int nl = lane & 15, hsel = lane >> 4;
  const int bh = blockIdx.x, b = bh >> 3, h = bh & 7;
  const int qbase = blockIdx.y * 128 + wave * 16;

#if HAS_TDM
  if (wave == 0) {
    tdm_copy_lds(&kps[0][0], kpT + (size_t)bh * PP * HD, PP * HD * 2);
    tdm_copy_lds(&vps[0][0], vp + (size_t)bh * HD * PP, HD * PP * 2);
    __builtin_amdgcn_s_wait_tensorcnt(0);
  }
#else
  {
    const float4* kg = (const float4*)(kpT + (size_t)bh * PP * HD);
    const float4* vg = (const float4*)(vp + (size_t)bh * HD * PP);
    float4* kd = (float4*)&kps[0][0];
    float4* vd = (float4*)&vps[0][0];
    for (int i = t; i < 2048; i += 256) { kd[i] = kg[i]; vd[i] = vg[i]; }
  }
#endif
  __syncthreads();

  const float tscale = temp[h];

  const float* __restrict__ qp = q + (size_t)(b * NN1 + qbase + nl) * CC + h * HD;
  const int kh = hsel * 8;
  const v16bf bq0 = cat8(ld8(qp + kh), ld8(qp + 16 + kh));
  const v16bf bq1 = cat8(ld8(qp + 32 + kh), ld8(qp + 48 + kh));

  v8f acc[16];
#pragma unroll
  for (int mt = 0; mt < 16; ++mt) {
    const int prow = mt * 16 + nl;
    const v16bf a0 = cat8(kps[prow][hsel], kps[prow][2 + hsel]);
    const v16bf a1 = cat8(kps[prow][4 + hsel], kps[prow][6 + hsel]);
    v8f c = {};
    c = wmma_bf16(a0, bq0, c);
    c = wmma_bf16(a1, bq1, c);
#pragma unroll
    for (int r = 0; r < 8; ++r) c[r] *= tscale;
    acc[mt] = c;
  }

  float mx = acc[0][0];
#pragma unroll
  for (int mt = 0; mt < 16; ++mt)
#pragma unroll
    for (int r = 0; r < 8; ++r) mx = fmaxf(mx, acc[mt][r]);
  mx = fmaxf(mx, __shfl_xor(mx, 16, 32));
  float sum = 0.f;
#pragma unroll
  for (int mt = 0; mt < 16; ++mt)
#pragma unroll
    for (int r = 0; r < 8; ++r) {
      const float e = __expf(acc[mt][r] - mx);
      acc[mt][r] = e;
      sum += e;
    }
  sum += __shfl_xor(sum, 16, 32);
  const float inv = 1.0f / sum;

  v16bf pf[8];
#pragma unroll
  for (int kc = 0; kc < 8; ++kc)
#pragma unroll
    for (int j = 0; j < 8; ++j) {
      pf[kc][j]     = (__bf16)(acc[2 * kc][j] * inv);
      pf[kc][8 + j] = (__bf16)(acc[2 * kc + 1][j] * inv);
    }

  v8f xacc[4] = {};
#pragma unroll
  for (int kc = 0; kc < 8; ++kc)
#pragma unroll
    for (int mt = 0; mt < 4; ++mt) {
      const int drow = mt * 16 + nl;
      const v16bf a = cat8(vps[drow][kc * 4 + hsel], vps[drow][kc * 4 + 2 + hsel]);
      xacc[mt] = wmma_bf16(a, pf[kc], xacc[mt]);
    }

  // reference transpose(0,3,1,2) flattening: off = b*N1*C + d*H*N1 + h*N1 + q
  const int qcol = qbase + nl;
  const int rb2 = hsel * 8;
#pragma unroll
  for (int mt = 0; mt < 4; ++mt)
#pragma unroll
    for (int r = 0; r < 8; ++r) {
      const int d = mt * 16 + rb2 + r;
      const size_t off =
          (size_t)b * (NN1 * CC) + (size_t)d * (HH * NN1) + h * NN1 + qcol;
      xo[off] = xacc[mt][r] + q[off];
    }
}

// ---------------------------------------------------------------------------
// Row L2-norm (+ optional bf16 copy). In-place safe.
// ---------------------------------------------------------------------------
__launch_bounds__(256)
__global__ void l2norm_rows(const float* __restrict__ src, float* __restrict__ dst,
                            __bf16* __restrict__ dstb, int C) {
  __shared__ float red[8];
  const int row = blockIdx.x;
  const float* __restrict__ sr = src + (size_t)row * C;
  float ss = 0.f;
  for (int i = threadIdx.x; i < C; i += 256) { const float v = sr[i]; ss += v * v; }
  for (int d = 1; d < 32; d <<= 1) ss += __shfl_xor(ss, d, 32);
  const int wave = threadIdx.x >> 5, lane = threadIdx.x & 31;
  if (lane == 0) red[wave] = ss;
  __syncthreads();
  if (wave == 0) {
    float v = (lane < 8) ? red[lane] : 0.f;
    for (int d = 1; d < 8; d <<= 1) v += __shfl_xor(v, d, 32);
    if (lane == 0) red[0] = v;
  }
  __syncthreads();
  const float invn = 1.0f / fmaxf(sqrtf(red[0]), 1e-12f);
  float* __restrict__ dr = dst + (size_t)row * C;
  for (int i = threadIdx.x; i < C; i += 256) {
    const float v = sr[i] * invn;
    dr[i] = v;
    if (dstb) dstb[(size_t)row * C + i] = (__bf16)v;
  }
}

// ---------------------------------------------------------------------------
extern "C" void kernel_launch(void* const* d_in, const int* in_sizes, int n_in,
                              void* d_out, int out_size, void* d_ws, size_t ws_size,
                              hipStream_t stream) {
  const float* x1   = (const float*)d_in[0];
  const float* x2   = (const float*)d_in[1];
  const float* Wq   = (const float*)d_in[2];
  const float* bq   = (const float*)d_in[3];
  const float* Wkv  = (const float*)d_in[4];
  const float* bkv  = (const float*)d_in[5];
  const float* Wp   = (const float*)d_in[6];
  const float* bp   = (const float*)d_in[7];
  const float* posq = (const float*)d_in[8];
  const float* posk = (const float*)d_in[9];
  const float* temp = (const float*)d_in[10];
  const float* W1   = (const float*)d_in[11];
  const float* b1   = (const float*)d_in[12];
  const float* W2   = (const float*)d_in[13];
  const float* b2   = (const float*)d_in[14];
  float* out = (float*)d_out;

  const int M = BB * NN1;  // 32768

  // ---- workspace layout (aliased lifetimes, peak ~300 MB) ----
  char* ws = (char*)d_ws;
  const size_t F32R = (size_t)M * CC * 4;   // 64 MB
  const size_t BFR  = (size_t)M * CC * 2;   // 32 MB
  float*  q_f32 = (float*)ws;               // region A: q, reused as xnorm
  float*  xnorm = (float*)ws;
  char* rb = ws + F32R;                     // region B: x1,x2,kT,vT -> h
  __bf16* x1_bf = (__bf16*)(rb);
  __bf16* x2_bf = (__bf16*)(rb + BFR);
  __bf16* kT_bf = (__bf16*)(rb + 2 * BFR);
  __bf16* vT_bf = (__bf16*)(rb + 3 * BFR);
  __bf16* h_bf  = (__bf16*)(rb);
  char* rc = rb + 4 * BFR;
  __bf16* WqT  = (__bf16*)(rc);                 rc += (size_t)CC * CC * 2;
  __bf16* WkvT = (__bf16*)(rc);                 rc += (size_t)CC * 2 * CC * 2;
  __bf16* WpT  = (__bf16*)(rc);                 rc += (size_t)NN2 * PP * 2;
  __bf16* W1T  = (__bf16*)(rc);                 rc += (size_t)CC * MLPD * 2;
  __bf16* W2T  = (__bf16*)(rc);                 rc += (size_t)MLPD * CC * 2;
  __bf16* kpT_bf = (__bf16*)(rc);               rc += (size_t)64 * HD * PP * 2;
  __bf16* vp_bf  = (__bf16*)(rc);               rc += (size_t)64 * HD * PP * 2;
  float*  xattn  = (float*)(rc);                rc += F32R;
  __bf16* xnorm_bf = (__bf16*)(rc);

  const dim3 blk(256);
  const int nelem = M * CC;  // 16.7M

  // 0) bf16 conversions / weight transposes
  cvt_bf16<<<dim3(nelem / 8 / 256), blk, 0, stream>>>(x1, x1_bf, nelem);
  cvt_bf16<<<dim3(nelem / 8 / 256), blk, 0, stream>>>(x2, x2_bf, nelem);
  transpose_cvt<<<dim3((CC * CC + 255) / 256), blk, 0, stream>>>(Wq, WqT, CC, CC);
  transpose_cvt<<<dim3((CC * 2 * CC + 255) / 256), blk, 0, stream>>>(Wkv, WkvT, CC, 2 * CC);
  transpose_cvt<<<dim3((NN2 * PP + 255) / 256), blk, 0, stream>>>(Wp, WpT, NN2, PP);
  transpose_cvt<<<dim3((CC * MLPD + 255) / 256), blk, 0, stream>>>(W1, W1T, CC, MLPD);
  transpose_cvt<<<dim3((MLPD * CC + 255) / 256), blk, 0, stream>>>(W2, W2T, MLPD, CC);

  // 1) q = x1 @ Wq + bq + pos_q (f32, kept for skip)
  gemm_wmma<MODE_Q><<<dim3(CC / 64, M / 128), blk, 0, stream>>>(
      x1_bf, WqT, bq, posq, q_f32, nullptr, nullptr, M, CC, CC, NN1);

  // 2) kv = x2 @ Wkv + bkv ; k += pos_k ; packed stores to kT/vT [b][h][d][n]
  gemm_wmma<MODE_KV><<<dim3(2 * CC / 64, M / 128), blk, 0, stream>>>(
      x2_bf, WkvT, bkv, posk, nullptr, kT_bf, vT_bf, M, 2 * CC, CC, NN2);

  // 3) kpT = (K^T @ Wp + bp)^T ; vp = V^T @ Wp + bp
  kpvp_wmma<<<dim3(BB * HH, PP / 64, 2), blk, 0, stream>>>(
      kT_bf, vT_bf, WpT, bp, kpT_bf, vp_bf);

  // 4) fused transposed softmax attention + q-skip (TDM tile loads)
  attn_fused<<<dim3(BB * HH, NN1 / 128), blk, 0, stream>>>(
      q_f32, kpT_bf, vp_bf, temp, xattn);

  // 5) xnorm = l2norm(xattn)  (+ bf16 copy for MLP1 A-operand)
  l2norm_rows<<<dim3(M), blk, 0, stream>>>(xattn, xnorm, xnorm_bf, CC);

  // 6) h = gelu(xnorm @ W1 + b1) -> bf16
  gemm_wmma<MODE_MLP1><<<dim3(MLPD / 64, M / 128), blk, 0, stream>>>(
      xnorm_bf, W1T, b1, nullptr, nullptr, h_bf, nullptr, M, MLPD, CC, 1);

  // 7) pre = h @ W2 + b2 + xnorm -> d_out
  gemm_wmma<MODE_MLP2><<<dim3(CC / 64, M / 128), blk, 0, stream>>>(
      h_bf, W2T, b2, xnorm, out, nullptr, nullptr, M, CC, MLPD, 1);

  // 8) out = l2norm(pre) in-place
  l2norm_rows<<<dim3(M), blk, 0, stream>>>(out, out, nullptr, CC);

  (void)in_sizes; (void)n_in; (void)out_size; (void)ws_size;
}
